// NeRF_34668976013427
// MI455X (gfx1250) — matrix operand order repaired and checked by reading, not compiled, not run on hardware
//
#include <hip/hip_runtime.h>
#include <hip/hip_bf16.h>

typedef __attribute__((ext_vector_type(16))) __bf16 v16bf;
typedef __attribute__((ext_vector_type(8)))  __bf16 v8bf;
typedef __attribute__((ext_vector_type(8)))  float  v8f;

#define N_PTS   32768
#define IN_CH   63
#define IN_CH_V 27
#define WID     256
#define D_INNER 512
#define D_STATE 16
#define DT_RANK 16
#define B_SEQ   8
#define L_SEQ   4096

union FragBF { v16bf v; v8bf h[2]; };

// =====================================================================
// Generic tiled bf16-WMMA GEMM:  C = act( A[M,K] @ W[K,N] + bias (+C) )
// Block = 256 threads (8 wave32), tile 128x128, BK=32.
// Wave (wr,wc) owns a 32x64 sub-tile = 2x4 WMMA 16x16 accumulators.
// A tile row-major in LDS; W tile stored TRANSPOSED ([n][k]) so both
// fragment types are 2 x 16B contiguous reads (ds_load_b128).
// Staging is two-phase (all global loads issued, then cvt+ds_store) so
// the 8 tile loads overlap instead of serializing on loadcnt==0.
// REQUIRES: M multiple of 128 and grid.x == M/128 (true for all calls).
// =====================================================================
template<int ACT, bool ACC>
__global__ __launch_bounds__(256) void gemm_wmma_bf16(
    const float* __restrict__ A, int lda,
    const float* __restrict__ W, int ldw,
    const float* __restrict__ bias,
    float* __restrict__ C, int ldc,
    int M, int N, int K)
{
  __shared__ __bf16 As[128][40];   // [m][k]
  __shared__ __bf16 Bs[128][40];   // [n][k]  (transposed W tile)

  const int tid  = threadIdx.x;
  const int lane = tid & 31;
  const int wave = tid >> 5;
  const int wr   = wave & 3;       // row strip (4 x 32 rows)
  const int wc   = wave >> 2;      // col strip (2 x 64 cols)
  const int half = lane >> 4;      // lane group 0/1
  const int l16  = lane & 15;
  const int rowBase = blockIdx.x * 128;
  const int colBase = blockIdx.y * 128;
  const bool fullN = (colBase + 128 <= N);

  // per-thread staging coordinates (constant across K loop)
  const int ar = tid >> 1;                 // A: row 0..127
  const int ac = (tid & 1) * 16;           // A: col 0 or 16  (2 x 16 floats? no)
  // A tile 128x32 = 4096 floats, 256 thr -> 16 floats = 4 runs of 4
  // run i: slot = tid + i*256 ; r = slot>>3 ; c = (slot&7)*4
  // B tile 32x128: run i: slot = tid + i*256 ; r = slot>>5 ; c = (slot&31)*4

  v8f acc[2][4];
  #pragma unroll
  for (int i = 0; i < 2; i++)
    #pragma unroll
    for (int j = 0; j < 4; j++)
      #pragma unroll
      for (int v = 0; v < 8; v++) acc[i][j][v] = 0.f;

  for (int k0 = 0; k0 < K; k0 += 32) {
    const bool fullK = (k0 + 32 <= K);

    float ta[4][4];   // A staging values
    float tb[4][4];   // W staging values

    // ---------- phase 1: issue ALL global loads ----------
    if (fullK) {
      #pragma unroll
      for (int i = 0; i < 4; i++) {
        int slot = tid + i * 256;
        int r = slot >> 3, c = (slot & 7) * 4;
        const float* p = A + (size_t)(rowBase + r) * lda + (k0 + c);
        #pragma unroll
        for (int j = 0; j < 4; j++) ta[i][j] = p[j];
      }
    } else {
      #pragma unroll
      for (int i = 0; i < 4; i++) {
        int slot = tid + i * 256;
        int r = slot >> 3, c = (slot & 7) * 4;
        const float* p = A + (size_t)(rowBase + r) * lda;
        #pragma unroll
        for (int j = 0; j < 4; j++) {
          int gk = k0 + c + j;
          float v = p[gk < K ? gk : (K - 1)];   // branchless clamp
          ta[i][j] = (gk < K) ? v : 0.f;
        }
      }
    }
    if (fullK && fullN) {
      #pragma unroll
      for (int i = 0; i < 4; i++) {
        int slot = tid + i * 256;
        int r = slot >> 5, c = (slot & 31) * 4;
        const float* p = W + (size_t)(k0 + r) * ldw + (colBase + c);
        #pragma unroll
        for (int j = 0; j < 4; j++) tb[i][j] = p[j];
      }
    } else {
      #pragma unroll
      for (int i = 0; i < 4; i++) {
        int slot = tid + i * 256;
        int r = slot >> 5, c = (slot & 31) * 4;
        int gk = k0 + r;
        int sk = gk < K ? gk : (K - 1);
        #pragma unroll
        for (int j = 0; j < 4; j++) {
          int gc = colBase + c + j;
          int sc = gc < N ? gc : (N - 1);
          float v = W[(size_t)sk * ldw + sc];
          tb[i][j] = (gk < K && gc < N) ? v : 0.f;
        }
      }
    }

    // ---------- phase 2: convert + LDS stores ----------
    #pragma unroll
    for (int i = 0; i < 4; i++) {
      int slot = tid + i * 256;
      int r = slot >> 3, c = (slot & 7) * 4;
      #pragma unroll
      for (int j = 0; j < 4; j++) As[r][c + j] = (__bf16)ta[i][j];
    }
    #pragma unroll
    for (int i = 0; i < 4; i++) {
      int slot = tid + i * 256;
      int r = slot >> 5, c = (slot & 31) * 4;
      #pragma unroll
      for (int j = 0; j < 4; j++) Bs[c + j][r] = (__bf16)tb[i][j];   // transpose
    }
    __syncthreads();

    // ---- fragments: each = 2 x 16B contiguous LDS reads ----
    // A 16x32 bf16 (ISA 7.12.2): lane-half h, row m=l16:
    //   elems 0..7 = K 8h..8h+7, elems 8..15 = K 16+8h..16+8h+7
    FragBF afr[2], bfr[4];
    #pragma unroll
    for (int tr = 0; tr < 2; tr++) {
      int m = wr * 32 + tr * 16 + l16;
      afr[tr].h[0] = *(const v8bf*)&As[m][8 * half];
      afr[tr].h[1] = *(const v8bf*)&As[m][16 + 8 * half];
    }
    // B 32x16 bf16: lane-half h, col n=l16: elems 0..7 = K 8h..8h+7, elems 8..15 = K 16+8h..16+8h+7
    #pragma unroll
    for (int tc = 0; tc < 4; tc++) {
      int n = wc * 64 + tc * 16 + l16;
      bfr[tc].h[0] = *(const v8bf*)&Bs[n][8 * half];
      bfr[tc].h[1] = *(const v8bf*)&Bs[n][16 + 8 * half];
    }
    #pragma unroll
    for (int tr = 0; tr < 2; tr++)
      #pragma unroll
      for (int tc = 0; tc < 4; tc++)
        acc[tr][tc] = __builtin_amdgcn_wmma_f32_16x16x32_bf16(
            false, afr[tr].v, false, bfr[tc].v, (short)0, acc[tr][tc], false, false);
    __syncthreads();
  }

  // ---- epilogue: C layout VGPR v -> row v + 8*half, col = lane&15 ----
  #pragma unroll
  for (int tr = 0; tr < 2; tr++) {
    #pragma unroll
    for (int tc = 0; tc < 4; tc++) {
      int col = colBase + wc * 64 + tc * 16 + l16;
      if (col < N) {
        float bv = bias ? bias[col] : 0.f;
        #pragma unroll
        for (int v = 0; v < 8; v++) {
          int row = rowBase + wr * 32 + tr * 16 + 8 * half + v;
          float val = acc[tr][tc][v] + bv;
          if (ACC) val += C[(size_t)row * ldc + col];
          if (ACT == 1) val = fmaxf(val, 0.f);
          C[(size_t)row * ldc + col] = val;
        }
      }
    }
  }
}

// =====================================================================
// Depthwise causal conv (K=4, left-pad 3) + bias + SiLU.
// =====================================================================
__global__ __launch_bounds__(256) void conv_silu_kernel(
    const float* __restrict__ xz, const float* __restrict__ conv_w,
    const float* __restrict__ conv_b, float* __restrict__ xm)
{
  int id = blockIdx.x * 256 + threadIdx.x;       // over N_PTS * D_INNER
  int d = id & (D_INNER - 1);
  int t = id >> 9;
  int l = t & (L_SEQ - 1);
  float w0 = conv_w[d * 4 + 0], w1 = conv_w[d * 4 + 1];
  float w2 = conv_w[d * 4 + 2], w3 = conv_w[d * 4 + 3];
  float s = conv_b[d] + w3 * xz[(size_t)t * 1024 + d];
  if (l >= 1) s += w2 * xz[(size_t)(t - 1) * 1024 + d];
  if (l >= 2) s += w1 * xz[(size_t)(t - 2) * 1024 + d];
  if (l >= 3) s += w0 * xz[(size_t)(t - 3) * 1024 + d];
  xm[(size_t)t * 512 + d] = s / (1.f + __expf(-s));   // silu
}

// =====================================================================
// Fused Mamba selective scan. One thread per (batch, channel).
// delta = softplus(dt @ dt_proj_w + b) in registers (K=16),
// 16-state recurrence in registers, y += u*D, y *= silu(z).
// Gated y overwrites z in place (xz col 512+d).
// =====================================================================
__global__ __launch_bounds__(128) void mamba_scan_kernel(
    const float* __restrict__ xm,     // u  [N_PTS, 512]
    float* __restrict__ xz,           // z in col 512+d; y overwrites it
    const float* __restrict__ dbc,    // [N_PTS, 48] = dt|B|C
    const float* __restrict__ dt_w,   // [16, 512]
    const float* __restrict__ dt_b,   // [512]
    const float* __restrict__ A_log,  // [512, 16]
    const float* __restrict__ Dp)     // [512]
{
  const int tid = threadIdx.x;
  const int d = blockIdx.x * 128 + tid;
  const int b = blockIdx.y;
  __shared__ float sdbc[64][49];

  float wd[16], Av[16], hst[16];
  #pragma unroll
  for (int r = 0; r < 16; r++) wd[r] = dt_w[r * 512 + d];
  #pragma unroll
  for (int s = 0; s < 16; s++) { Av[s] = -__expf(A_log[d * 16 + s]); hst[s] = 0.f; }
  const float dtb = dt_b[d];
  const float Dd  = Dp[d];

  for (int l0 = 0; l0 < L_SEQ; l0 += 64) {
    __syncthreads();
    const size_t base = (size_t)b * L_SEQ + l0;
    #pragma unroll
    for (int i = 0; i < 24; i++) {        // 64*48 / 128 threads
      int idx = tid + i * 128;
      int j = idx / 48, c = idx % 48;
      sdbc[j][c] = dbc[(base + j) * 48 + c];
    }
    __syncthreads();
    for (int j = 0; j < 64; j++) {
      size_t t = base + j;
      float u  = xm[t * 512 + d];
      float zv = xz[t * 1024 + 512 + d];
      float a = dtb;
      #pragma unroll
      for (int r = 0; r < 16; r++) a += sdbc[j][r] * wd[r];
      float delta = (a > 20.f) ? a : log1pf(__expf(a));   // softplus
      float du = delta * u;
      float y = 0.f;
      #pragma unroll
      for (int s = 0; s < 16; s++) {
        float dA = __expf(delta * Av[s]);
        hst[s] = dA * hst[s] + du * sdbc[j][16 + s];
        y += hst[s] * sdbc[j][32 + s];
      }
      y += u * Dd;
      float sz = zv / (1.f + __expf(-zv));                // silu(z)
      xz[t * 1024 + 512 + d] = y * sz;
    }
  }
}

// ======================= alpha head: N=1 dot ==========================
__global__ __launch_bounds__(256) void alpha_kernel(
    const float* __restrict__ h2, const float* __restrict__ w_a,
    const float* __restrict__ b_a, float* __restrict__ out)
{
  int lane = threadIdx.x & 31;
  int wave = threadIdx.x >> 5;
  int t = blockIdx.x * 8 + wave;
  float v = 0.f;
  #pragma unroll
  for (int i = 0; i < 8; i++) {
    int k = lane + 32 * i;
    v += h2[(size_t)t * 256 + k] * w_a[k];
  }
  #pragma unroll
  for (int off = 16; off > 0; off >>= 1) v += __shfl_down(v, off, 32);
  if (lane == 0) out[(size_t)t * 4 + 3] = v + b_a[0];
}

// ======================= rgb head: N=3 ================================
__global__ __launch_bounds__(256) void rgb_kernel(
    const float* __restrict__ hv, const float* __restrict__ w_rgb,
    const float* __restrict__ b_rgb, float* __restrict__ out)
{
  __shared__ float wsm[128 * 3];
  int tid = threadIdx.x;
  if (tid < 128 * 3) wsm[tid] = w_rgb[tid];
  __syncthreads();
  int t = blockIdx.x * 256 + tid;
  float a0 = b_rgb[0], a1 = b_rgb[1], a2 = b_rgb[2];
  #pragma unroll 4
  for (int k = 0; k < 128; k++) {
    float v = hv[(size_t)t * 128 + k];
    a0 += v * wsm[k * 3 + 0];
    a1 += v * wsm[k * 3 + 1];
    a2 += v * wsm[k * 3 + 2];
  }
  out[(size_t)t * 4 + 0] = a0;
  out[(size_t)t * 4 + 1] = a1;
  out[(size_t)t * 4 + 2] = a2;
}

// =====================================================================
extern "C" void kernel_launch(void* const* d_in, const int* in_sizes, int n_in,
                              void* d_out, int out_size, void* d_ws, size_t ws_size,
                              hipStream_t stream)
{
  (void)in_sizes; (void)n_in; (void)out_size; (void)ws_size;
  const float* x        = (const float*)d_in[0];   // [32768, 90]
  const float* w_pts0   = (const float*)d_in[1];   // [63, 256]
  const float* b_pts0   = (const float*)d_in[2];
  const float* in_proj  = (const float*)d_in[3];   // [256, 1024]
  const float* conv_w   = (const float*)d_in[4];   // [512, 4]
  const float* conv_b   = (const float*)d_in[5];
  const float* x_proj   = (const float*)d_in[6];   // [512, 48]
  const float* dt_w     = (const float*)d_in[7];   // [16, 512]
  const float* dt_b     = (const float*)d_in[8];
  const float* A_log    = (const float*)d_in[9];   // [512, 16]
  const float* D_param  = (const float*)d_in[10];
  const float* out_proj = (const float*)d_in[11];  // [512, 256]
  const float* w_feat   = (const float*)d_in[12];  // [256, 256]
  const float* b_feat   = (const float*)d_in[13];
  const float* w_alpha  = (const float*)d_in[14];  // [256, 1]
  const float* b_alpha  = (const float*)d_in[15];
  const float* w_views  = (const float*)d_in[16];  // [283, 128]
  const float* b_views  = (const float*)d_in[17];
  const float* w_rgb    = (const float*)d_in[18];  // [128, 3]
  const float* b_rgb    = (const float*)d_in[19];
  float* out = (float*)d_out;

  float* ws = (float*)d_ws;
  size_t off = 0;
  float* buf_h   = ws + off; off += (size_t)N_PTS * WID;    // pre-mamba h, later h2
  float* buf_xz  = ws + off; off += (size_t)N_PTS * 1024;   // xz; z-half becomes y; later feature
  float* buf_xm  = ws + off; off += (size_t)N_PTS * 512;    // conv output u; later hv
  float* buf_dbc = ws + off; off += (size_t)N_PTS * 48;     // dt|B|C
  float* buf_feat = buf_xz;   // reuse after scan/out_proj
  float* buf_hv   = buf_xm;   // reuse after out_proj

  dim3 blk(256);

  // 1) h = pts @ w_pts0 + b           (M=32768, K=63, N=256)
  gemm_wmma_bf16<0,false><<<dim3(256, 2), blk, 0, stream>>>(
      x, IN_CH + IN_CH_V, w_pts0, WID, b_pts0, buf_h, WID, N_PTS, WID, IN_CH);

  // 2) xz = h @ in_proj_w             (K=256, N=1024)
  gemm_wmma_bf16<0,false><<<dim3(256, 8), blk, 0, stream>>>(
      buf_h, WID, in_proj, 1024, nullptr, buf_xz, 1024, N_PTS, 1024, WID);

  // 3) xm = silu(causal_dwconv(xz[:, :512]) + conv_b)
  conv_silu_kernel<<<dim3((N_PTS * D_INNER) / 256), blk, 0, stream>>>(
      buf_xz, conv_w, conv_b, buf_xm);

  // 4) dbc = xm @ x_proj_w            (K=512, N=48)
  gemm_wmma_bf16<0,false><<<dim3(256, 1), blk, 0, stream>>>(
      buf_xm, D_INNER, x_proj, 48, nullptr, buf_dbc, 48, N_PTS, 48, D_INNER);

  // 5) fused selective scan (+dt_proj, +u*D, *silu(z)); y -> xz z-half
  mamba_scan_kernel<<<dim3(4, B_SEQ), dim3(128), 0, stream>>>(
      buf_xm, buf_xz, buf_dbc, dt_w, dt_b, A_log, D_param);

  // 6) h2 = y @ out_proj_w            (A = xz+512, lda=1024, K=512, N=256)
  gemm_wmma_bf16<0,false><<<dim3(256, 2), blk, 0, stream>>>(
      buf_xz + 512, 1024, out_proj, WID, nullptr, buf_h, WID, N_PTS, WID, D_INNER);

  // 7) feature = h2 @ w_feat + b_feat (K=256, N=256)
  gemm_wmma_bf16<0,false><<<dim3(256, 2), blk, 0, stream>>>(
      buf_h, WID, w_feat, WID, b_feat, buf_feat, WID, N_PTS, WID, WID);

  // 8) alpha = h2 @ w_alpha + b_alpha -> out[:,3]
  alpha_kernel<<<dim3(N_PTS / 8), blk, 0, stream>>>(buf_h, w_alpha, b_alpha, out);

  // 9) hv = relu(feature @ Wv[:256] + views @ Wv[256:283] + b_views)
  gemm_wmma_bf16<0,false><<<dim3(256, 1), blk, 0, stream>>>(
      buf_feat, WID, w_views, 128, b_views, buf_hv, 128, N_PTS, 128, WID);
  gemm_wmma_bf16<1,true><<<dim3(256, 1), blk, 0, stream>>>(
      x + IN_CH, IN_CH + IN_CH_V, w_views + (size_t)WID * 128, 128, nullptr,
      buf_hv, 128, N_PTS, 128, IN_CH_V);

  // 10) rgb = hv @ w_rgb + b_rgb -> out[:,0:3]
  rgb_kernel<<<dim3(N_PTS / 256), blk, 0, stream>>>(buf_hv, w_rgb, b_rgb, out);
}
